// C2VQEmbedding_48885317763882
// MI455X (gfx1250) — matrix operand order repaired and checked by reading, not compiled, not run on hardware
//
#include <hip/hip_runtime.h>

// ---------------------------------------------------------------------------
// Conditional VQ nearest-code lookup for MI455X (gfx1250, wave32, WMMA).
//
// z_e_x: [B=32, D=256, H=64, W=64] f32   c: [B] int   embedding: [60*512, 256]
// out:   [B, H, W, D] f32  (gathered fp32 code rows)
//
// dist GEMM in bf16 WMMA (v_wmma_f32_16x16x32_bf16), f32 accumulate.
// argmin(dist) invariant to ||z||^2, so dist = ||cb||^2 - 2 z.cb.
// cb chunks are double-buffered in LDS: buffer 0 reuses the (dead after A-frag
// load) z transpose-staging region, so total LDS stays ~51KB.
// ---------------------------------------------------------------------------

typedef __attribute__((ext_vector_type(16))) __bf16 v16bf;
typedef __attribute__((ext_vector_type(8)))  __bf16 v8bf;
typedef __attribute__((ext_vector_type(8)))  float  v8f;

#define NUM_CLASSES 60
#define KCODES      512
#define DDIM        256
#define HW          4096       // 64*64
#define ROWS        64         // z rows per block
#define BLOCK       128        // 4 wave32s
#define PZ          264        // z LDS pitch (bf16 elems), multiple of 8
#define PC          264        // cb LDS pitch (bf16 elems), multiple of 8
#define NCHUNK      16         // 512 codes / 32 per chunk

__device__ __forceinline__ unsigned short f2bf(float f) {
    unsigned int u = __float_as_uint(f);
    unsigned int r = u + 0x7FFFu + ((u >> 16) & 1u);   // round-to-nearest-even
    return (unsigned short)(r >> 16);
}

__global__ __launch_bounds__(BLOCK)
void vq_wmma_kernel(const float* __restrict__ z,
                    const int*   __restrict__ c,
                    const float* __restrict__ emb,
                    float*       __restrict__ out)
{
    // Carved shared buffer:
    //   [0, 16896)        ushorts: z staging (phase 1-2) / cb buffer 0 (phase 3)
    //   [16896, 25344)    ushorts: cb buffer 1
    __shared__ __align__(16) unsigned short smem[16896 + 8448];
    __shared__ float cb_sq[2][32];
    __shared__ int   idx_lds[ROWS];

    unsigned short* const z_lds = smem;
    unsigned short* const cbuf0 = smem;             // reuses z region
    unsigned short* const cbuf1 = smem + 16896;

    const int tid  = threadIdx.x;
    const int lane = tid & 31;
    const int wv   = tid >> 5;          // wave 0..3
    const int half = lane >> 4;         // 0/1
    const int l15  = lane & 15;

    const int bid = blockIdx.x;         // 2048 blocks
    const int b   = bid >> 6;           // bid / 64
    const int n0  = (bid & 63) << 6;    // row-tile start within batch

    const int  cls    = c[b];
    const long cbBase = (long)cls * KCODES * DDIM;   // f32 element offset

    // ---- Phase 1: stage z tile [ROWS x DDIM] into LDS (transpose, f32->bf16)
    {
        const float* zb = z + (long)b * DDIM * HW + n0;
        const int quad = tid & 15;      // float4 index along n
        const int drow = tid >> 4;      // 0..7
        for (int pass = 0; pass < 32; ++pass) {
            const int d = pass * 8 + drow;
            const float4 v = *(const float4*)(zb + (long)d * HW + quad * 4);
            const int nl = quad * 4;
            z_lds[(nl + 0) * PZ + d] = f2bf(v.x);
            z_lds[(nl + 1) * PZ + d] = f2bf(v.y);
            z_lds[(nl + 2) * PZ + d] = f2bf(v.z);
            z_lds[(nl + 3) * PZ + d] = f2bf(v.w);
        }
    }
    __syncthreads();

    // ---- Phase 2: per-wave A fragments (16 z rows) resident in registers.
    // A frag k order used here (contiguous, B matches): lanes 0-15 hold K 0..15; lanes 16-31
    // hold K 16..31  -> two 16B chunks per fragment.
    v16bf afrag[8];
    {
        const unsigned short* zrow = &z_lds[(wv * 16 + l15) * PZ];
#pragma unroll
        for (int f = 0; f < 8; ++f) {
            const int d0 = f * 32;
            union { v16bf v; v8bf h[2]; } u;
            u.h[0] = *(const v8bf*)(zrow + d0 + half * 16);
            u.h[1] = *(const v8bf*)(zrow + d0 + half * 16 + 8);
            afrag[f] = u.v;
        }
    }
    __syncthreads();   // z region is now dead -> reusable as cb buffer 0

    float minv[8];
    int   mini[8];
#pragma unroll
    for (int v = 0; v < 8; ++v) { minv[v] = 3.4e38f; mini[v] = 0; }

    // cb chunk load mapping: thread -> code = tid>>2 (0..31), quarter = tid&3
    const int code = tid >> 2;
    const int part = tid & 3;

    // ---- Preload chunk 0 into buffer 0
    {
        const float* src = emb + cbBase + (long)code * DDIM + part * 64;
        unsigned short* dst = cbuf0 + code * PC + part * 64;
        float s = 0.f;
#pragma unroll
        for (int j = 0; j < 16; ++j) {
            const float4 vv = *(const float4*)(src + j * 4);
            s += vv.x * vv.x + vv.y * vv.y + vv.z * vv.z + vv.w * vv.w;
            dst[j * 4 + 0] = f2bf(vv.x);
            dst[j * 4 + 1] = f2bf(vv.y);
            dst[j * 4 + 2] = f2bf(vv.z);
            dst[j * 4 + 3] = f2bf(vv.w);
        }
        s += __shfl_xor(s, 1, 32);
        s += __shfl_xor(s, 2, 32);
        if (part == 0) cb_sq[0][code] = s;
    }
    __syncthreads();

    // ---- Phase 3: double-buffered chunk loop.
    // Per iteration: issue global loads for chunk kc+1 (half at a time),
    // run WMMAs for chunk kc, then convert+store kc+1 to the other buffer.
    for (int kc = 0; kc < NCHUNK; ++kc) {
        const int  nxt = kc + 1;
        const bool has = (nxt < NCHUNK);             // uniform across block
        const unsigned short* cur = (kc & 1) ? cbuf1 : cbuf0;
        unsigned short*       nbf = (nxt & 1) ? cbuf1 : cbuf0;
        const float* nsrc = emb + cbBase + (long)(nxt * 32 + code) * DDIM + part * 64;
        unsigned short* ndst = nbf + code * PC + part * 64;

        float  sq = 0.f;
        float4 pf[8];

        // -- prefetch half 0 of next chunk (32 floats/thread)
        if (has) {
#pragma unroll
            for (int j = 0; j < 8; ++j) pf[j] = *(const float4*)(nsrc + j * 4);
        }

        // -- compute column tile 0 (codes kc*32 .. +15)
        {
            v8f acc = {};
            const unsigned short* crow = cur + l15 * PC;
#pragma unroll
            for (int f = 0; f < 8; ++f) {
                const int d0 = f * 32 + half * 16;
                union { v16bf v; v8bf h[2]; } u;
                u.h[0] = *(const v8bf*)(crow + d0);
                u.h[1] = *(const v8bf*)(crow + d0 + 8);
                acc = __builtin_amdgcn_wmma_f32_16x16x32_bf16(
                        false, afrag[f], false, u.v, (short)0, acc, false, false);
            }
            const float csq = cb_sq[kc & 1][l15];
            const int   ci  = kc * 32 + l15;
#pragma unroll
            for (int v = 0; v < 8; ++v) {
                const float dist = csq - 2.0f * acc[v];
                if (dist < minv[v]) { minv[v] = dist; mini[v] = ci; }
            }
        }

        // -- store half 0, prefetch half 1
        if (has) {
#pragma unroll
            for (int j = 0; j < 8; ++j) {
                const float4 vv = pf[j];
                sq += vv.x * vv.x + vv.y * vv.y + vv.z * vv.z + vv.w * vv.w;
                ndst[j * 4 + 0] = f2bf(vv.x);
                ndst[j * 4 + 1] = f2bf(vv.y);
                ndst[j * 4 + 2] = f2bf(vv.z);
                ndst[j * 4 + 3] = f2bf(vv.w);
            }
#pragma unroll
            for (int j = 0; j < 8; ++j) pf[j] = *(const float4*)(nsrc + 32 + j * 4);
        }

        // -- compute column tile 1 (codes kc*32+16 .. +31)
        {
            v8f acc = {};
            const unsigned short* crow = cur + (16 + l15) * PC;
#pragma unroll
            for (int f = 0; f < 8; ++f) {
                const int d0 = f * 32 + half * 16;
                union { v16bf v; v8bf h[2]; } u;
                u.h[0] = *(const v8bf*)(crow + d0);
                u.h[1] = *(const v8bf*)(crow + d0 + 8);
                acc = __builtin_amdgcn_wmma_f32_16x16x32_bf16(
                        false, afrag[f], false, u.v, (short)0, acc, false, false);
            }
            const float csq = cb_sq[kc & 1][16 + l15];
            const int   ci  = kc * 32 + 16 + l15;
#pragma unroll
            for (int v = 0; v < 8; ++v) {
                const float dist = csq - 2.0f * acc[v];
                if (dist < minv[v]) { minv[v] = dist; mini[v] = ci; }
            }
        }

        // -- store half 1 + finish next chunk's ||cb||^2
        if (has) {
#pragma unroll
            for (int j = 0; j < 8; ++j) {
                const float4 vv = pf[j];
                sq += vv.x * vv.x + vv.y * vv.y + vv.z * vv.z + vv.w * vv.w;
                ndst[32 + j * 4 + 0] = f2bf(vv.x);
                ndst[32 + j * 4 + 1] = f2bf(vv.y);
                ndst[32 + j * 4 + 2] = f2bf(vv.z);
                ndst[32 + j * 4 + 3] = f2bf(vv.w);
            }
            sq += __shfl_xor(sq, 1, 32);
            sq += __shfl_xor(sq, 2, 32);
            if (part == 0) cb_sq[nxt & 1][code] = sq;
        }

        __syncthreads();   // next chunk visible; current buffer reusable
    }

    // ---- Phase 4: argmin reduce across the 16 columns (lanes) of each half.
    // C layout: VGPR v <-> row (v + 8*half), column = l15.
#pragma unroll
    for (int v = 0; v < 8; ++v) {
        float mv = minv[v];
        int   mi = mini[v];
#pragma unroll
        for (int m = 8; m >= 1; m >>= 1) {
            const float ov = __shfl_xor(mv, m, 32);
            const int   oi = __shfl_xor(mi, m, 32);
            if (ov < mv || (ov == mv && oi < mi)) { mv = ov; mi = oi; }
        }
        if (l15 == 0) idx_lds[wv * 16 + half * 8 + v] = mi;
    }
    __syncthreads();

    // ---- Phase 5: gather winning fp32 code rows to out[b][h][w][d]
    float* ob = out + ((long)b * HW + n0) * DDIM;
    for (int r = wv; r < ROWS; r += 4) {
        const int idx = idx_lds[r];
        const float4* src = (const float4*)(emb + cbBase + (long)idx * DDIM);
        float4*       dst = (float4*)(ob + (long)r * DDIM);
#pragma unroll
        for (int j = 0; j < 2; ++j)
            dst[lane + j * 32] = src[lane + j * 32];
    }
}

extern "C" void kernel_launch(void* const* d_in, const int* in_sizes, int n_in,
                              void* d_out, int out_size, void* d_ws, size_t ws_size,
                              hipStream_t stream) {
    (void)in_sizes; (void)n_in; (void)out_size; (void)d_ws; (void)ws_size;
    const float* z   = (const float*)d_in[0];
    const int*   c   = (const int*)d_in[1];
    const float* emb = (const float*)d_in[2];
    float*       out = (float*)d_out;

    const int nBlocks = 32 * (HW / ROWS);   // 2048
    hipLaunchKernelGGL(vq_wmma_kernel, dim3(nBlocks), dim3(BLOCK), 0, stream,
                       z, c, emb, out);
}
